// HNSWBlock_38139309588849
// MI455X (gfx1250) — matrix operand order repaired and checked by reading, not compiled, not run on hardware
//
#include <hip/hip_runtime.h>
#include <hip/hip_bf16.h>
#include <math.h>

#define DEVI __device__ __forceinline__

typedef __attribute__((ext_vector_type(16))) __bf16 v16bf;
typedef __attribute__((ext_vector_type(8)))  __bf16 v8bf;
typedef __attribute__((ext_vector_type(8)))  float  v8f;
typedef __attribute__((ext_vector_type(4)))  unsigned int v4u;
typedef __attribute__((ext_vector_type(8)))  int v8i;
typedef __attribute__((ext_vector_type(4)))  int v4i;

constexpr int N_ = 32, C_ = 2048, K_ = 16, B_ = 4;

DEVI float bf2f(__hip_bfloat16 h){ return __bfloat162float(h); }
DEVI float lo_bf(unsigned int u){ return __uint_as_float(u << 16); }
DEVI float hi_bf(unsigned int u){ return __uint_as_float(u & 0xFFFF0000u); }

DEVI float wave_sum(float v){
  #pragma unroll
  for (int o = 16; o > 0; o >>= 1) v += __shfl_xor(v, o, 32);
  return v;
}

DEVI float residual_scale(int level){
  float th = 1.04719755f;            // pi/3
  float p = 1.f;
  for (int l = 0; l < level; ++l) p *= 1.5f;
  th = th / p;
  float c = __cosf(th);
  return sqrtf((1.f/(c*c) - 1.f) * (1.f/(float)C_));
}

// ---- TDM: stage one row of `nelem` bf16 from global into LDS (async) ----
DEVI void tdm_load_row_bf16(unsigned int lds_addr, const void* gsrc, int nelem){
  unsigned long long ga = (unsigned long long)(uintptr_t)gsrc;
  v4u g0;
  g0[0] = 1u;                                            // count=1, user mode
  g0[1] = lds_addr;                                      // lds_addr [63:32]
  g0[2] = (unsigned int)(ga & 0xFFFFFFFFull);            // global_addr lo
  g0[3] = (unsigned int)((ga >> 32) & 0x1FFFFFFull) | (2u << 30); // addr hi + type=2
  v8i g1;
  g1[0] = (1 << 16);                                     // data_size = 2 bytes
  g1[1] = (int)(((unsigned)nelem) << 16);                // tensor_dim0 [79:48] lo
  g1[2] = (int)((((unsigned)nelem) >> 16) & 0xFFFFu) | (1 << 16); // dim0 hi, tensor_dim1=1
  g1[3] = (int)(((unsigned)nelem) << 16);                // tile_dim0 [127:112]
  g1[4] = 1;                                             // tile_dim1 = 1
  g1[5] = nelem;                                         // tensor_dim0_stride
  g1[6] = 0; g1[7] = 0;
  v4i z = {0,0,0,0};
#if defined(__clang_major__) && __clang_major__ >= 23
  v8i z8 = {0,0,0,0,0,0,0,0};
  __builtin_amdgcn_tensor_load_to_lds(g0, g1, z, z, z8, 0);
#else
  __builtin_amdgcn_tensor_load_to_lds(g0, g1, z, z, 0);
#endif
}

// ============ Phase 1: rmsnorm(x) + level-0 scores + top-16 ============
__global__ __launch_bounds__(256) void k_rms_score(
    const __hip_bfloat16* __restrict__ x, const float* __restrict__ nscale,
    const __hip_bfloat16* __restrict__ gate0,
    __hip_bfloat16* __restrict__ t_ws, int* __restrict__ idx0,
    float* __restrict__ acc)
{
  __shared__ __align__(64) __hip_bfloat16 sh_t[C_];
  __shared__ float red[8];
  __shared__ float sh_s[N_];
  __shared__ float bcast;
  int b = blockIdx.x;
  int tid = threadIdx.x, lane = tid & 31, w = tid >> 5;

  for (int c = tid; c < C_; c += 256) acc[b*C_ + c] = 0.f;   // zero out-accumulator

  float ss = 0.f;
  for (int c = tid; c < C_; c += 256) {
    float xf = bf2f(x[b*C_ + c]);
    ss += xf*xf;
  }
  ss = wave_sum(ss);
  if (lane == 0) red[w] = ss;
  __syncthreads();
  if (tid == 0) {
    float s = 0.f; for (int q = 0; q < 8; ++q) s += red[q];
    bcast = rsqrtf(s * (1.f/(float)C_) + 1e-5f);
  }
  __syncthreads();
  float inv = bcast;
  for (int c = tid; c < C_; c += 256) {
    float tf = bf2f(x[b*C_ + c]) * inv * nscale[c];
    __hip_bfloat16 tb = __float2bfloat16(tf);
    sh_t[c] = tb;
    t_ws[b*C_ + c] = tb;
  }
  __syncthreads();

  for (int rr = 0; rr < 4; ++rr) {                // 8 waves x 4 rows = 32 gate rows
    int n = w*4 + rr;
    const __hip_bfloat16* g = gate0 + (size_t)n*C_;
    float tg = 0.f, gg = 0.f;
    for (int c = lane; c < C_; c += 32) {
      float gf = bf2f(g[c]);
      tg += bf2f(sh_t[c]) * gf;
      gg += gf*gf;
    }
    tg = wave_sum(tg); gg = wave_sum(gg);
    if (lane == 0) sh_s[n] = tg / fmaxf(sqrtf(gg), 1e-12f);
  }
  __syncthreads();
  if (tid < 32) {                                  // wave-level top-16
    float v = sh_s[lane];
    for (int it = 0; it < K_; ++it) {
      float bv = v; int bi = lane;
      #pragma unroll
      for (int o = 16; o > 0; o >>= 1) {
        float ov = __shfl_xor(bv, o, 32); int oi = __shfl_xor(bi, o, 32);
        if (ov > bv || (ov == bv && oi < bi)) { bv = ov; bi = oi; }
      }
      if (lane == 0) idx0[b*K_ + it] = bi;
      if (lane == bi) v = -3.0e38f;
    }
  }
}

// ============ Phase 2: level-1 scores, top-16, materialize parents ============
__global__ __launch_bounds__(256) void k_level1(
    const __hip_bfloat16* __restrict__ gate0,
    const __hip_bfloat16* __restrict__ gate1,
    const __hip_bfloat16* __restrict__ t_ws,
    const int* __restrict__ idx0,
    int* __restrict__ idx1, float* __restrict__ tp_ws,
    __hip_bfloat16* __restrict__ p2)
{
  __shared__ __align__(64) __hip_bfloat16 sh_t[C_];
  __shared__ __align__(64) __hip_bfloat16 sh_g0[C_];
  __shared__ float red0[8], red1[8];
  __shared__ float sh_s1[N_], sh_inv[N_];
  __shared__ int   sh_sel[K_];
  __shared__ float sh_b[2];

  int b = blockIdx.x >> 4, i = blockIdx.x & 15;
  int tid = threadIdx.x, lane = tid & 31, w = tid >> 5;
  int ii = idx0[b*K_ + i];
  float sc1 = residual_scale(1);

  float tg = 0.f, gg = 0.f;
  for (int c = tid; c < C_; c += 256) {
    __hip_bfloat16 tv = t_ws[b*C_ + c];
    __hip_bfloat16 gv = gate0[(size_t)ii*C_ + c];
    sh_t[c] = tv; sh_g0[c] = gv;
    float tf = bf2f(tv), gf = bf2f(gv);
    tg += tf*gf; gg += gf*gf;
  }
  tg = wave_sum(tg); gg = wave_sum(gg);
  if (lane == 0) { red0[w] = tg; red1[w] = gg; }
  __syncthreads();
  if (tid == 0) {
    float a = 0.f, bq = 0.f;
    for (int q = 0; q < 8; ++q) { a += red0[q]; bq += red1[q]; }
    float invn = 1.f / fmaxf(sqrtf(bq), 1e-12f);
    sh_b[0] = invn; sh_b[1] = a * invn;    // 1/||g0||, t . p0
  }
  __syncthreads();
  float inv_ng0 = sh_b[0], tp0 = sh_b[1];

  for (int rr = 0; rr < 4; ++rr) {
    int n = w + 8*rr;
    const __hip_bfloat16* g = gate1 + ((size_t)ii*N_ + n)*C_;
    float a = 0.f, bg = 0.f, cc = 0.f;
    for (int c = lane; c < C_; c += 32) {
      float gf = bf2f(g[c]);
      a  += bf2f(sh_t[c])  * gf;
      bg += bf2f(sh_g0[c]) * gf;
      cc += gf*gf;
    }
    a = wave_sum(a); bg = wave_sum(bg); cc = wave_sum(cc);
    if (lane == 0) {
      float nn = sqrtf(fmaxf(1.f + 2.f*sc1*bg*inv_ng0 + sc1*sc1*cc, 0.f));
      float invq = 1.f / fmaxf(nn, 1e-12f);
      sh_s1[n]  = (tp0 + sc1*a) * invq;    // == t . l2norm(p0 + sc1*g1row)
      sh_inv[n] = invq;
    }
  }
  __syncthreads();
  if (tid < 32) {
    float v = sh_s1[lane];
    for (int it = 0; it < K_; ++it) {
      float bv = v; int bi = lane;
      #pragma unroll
      for (int o = 16; o > 0; o >>= 1) {
        float ov = __shfl_xor(bv, o, 32); int oi = __shfl_xor(bi, o, 32);
        if (ov > bv || (ov == bv && oi < bi)) { bv = ov; bi = oi; }
      }
      if (lane == 0) sh_sel[it] = bi;
      if (lane == bi) v = -3.0e38f;
    }
  }
  __syncthreads();
  if (tid < K_) {
    int j = sh_sel[tid];
    idx1 [(b*K_+i)*K_ + tid] = j;
    tp_ws[(b*K_+i)*K_ + tid] = sh_s1[j];   // t . parent2  (reused at leaf level)
  }
  for (int kk = 0; kk < K_; ++kk) {        // materialize normalized parent2 rows
    int j = sh_sel[kk];
    float invq = sh_inv[j];
    const __hip_bfloat16* g = gate1 + ((size_t)ii*N_ + j)*C_;
    __hip_bfloat16* dst = p2 + (((size_t)(b*K_+i))*K_ + kk)*C_;
    for (int c = tid; c < C_; c += 256) {
      float p0c = bf2f(sh_g0[c]) * inv_ng0;
      dst[c] = __float2bfloat16((p0c + sc1*bf2f(g[c])) * invq);
    }
  }
}

// ============ Phase 3: leaf level — WMMA triple-dots + weighted V accumulate ============
__global__ __launch_bounds__(256) void k_leaf(
    const __hip_bfloat16* __restrict__ mlp1,
    const __hip_bfloat16* __restrict__ mlp2,
    const __hip_bfloat16* __restrict__ t_ws,
    const __hip_bfloat16* __restrict__ p2,
    const int* __restrict__ idx0, const int* __restrict__ idx1,
    const float* __restrict__ tp_ws,
    const __hip_bfloat16* __restrict__ scale_p,
    float* __restrict__ acc)
{
  __shared__ __align__(64) __hip_bfloat16 sh_t[C_];
  __shared__ __align__(64) __hip_bfloat16 sh_p[C_];
  __shared__ __align__(64) __hip_bfloat16 sh_zero[32];
  __shared__ float r_t[2][4][16], r_p[2][4][16], r_s[2][4][16];
  __shared__ float sh_h[N_];

  int blk = blockIdx.x;
  int b = blk >> 8, i = (blk >> 4) & 15, j = blk & 15;
  int tid = threadIdx.x, lane = tid & 31, w = tid >> 5;
  int group = w >> 2, wq = w & 3, half = lane >> 4, li = lane & 15;

  int ii = idx0[b*K_ + i];
  int jj = idx1[(b*K_ + i)*K_ + j];
  float tp = tp_ws[(b*K_ + i)*K_ + j];
  float sc2 = residual_scale(2);
  float scale_f = bf2f(scale_p[0]);

  if (tid < 32) sh_zero[tid] = __float2bfloat16(0.0f);   // zero pad for B cols >= 2

  if (w == 0) {  // async TDM staging of t and parent rows into LDS
    tdm_load_row_bf16((unsigned int)(uintptr_t)(void*)sh_t, t_ws + (size_t)b*C_, C_);
    tdm_load_row_bf16((unsigned int)(uintptr_t)(void*)sh_p,
                      p2 + (((size_t)(b*K_+i))*K_ + j)*C_, C_);
    __builtin_amdgcn_s_wait_tensorcnt((short)0);
  }
  __syncthreads();

  // uniform scalar base + 32-bit per-lane element offsets (saddr+voffset form)
  const __hip_bfloat16* kbase = mlp1 + (((size_t)ii*N_ + jj)*N_)*C_;
  const unsigned rbase = (unsigned)((group*16 + li) * C_);   // this lane's leaf row

  // branchless B_tp source: lanes 0/1 -> t/p rows, others -> zero pad.
  // LDS offset is a pure induction variable: step is 0 for lanes >= 2.
  const __hip_bfloat16* bbase = (li == 0) ? sh_t : (li == 1) ? sh_p : sh_zero;
  const unsigned bmask = (li < 2) ? ~0u : 0u;
  const unsigned cA0 = half ? 16u : 0u, cA1 = half ? 24u : 8u;
  const unsigned cB  = (unsigned)(half * 16);
  unsigned btoff = ((unsigned)(wq * 32) + cB) & bmask;       // elements
  const unsigned btstep = 128u & bmask;                      // 4 waves * 32 k / chunk

  v8f d1 = {0.f,0.f,0.f,0.f,0.f,0.f,0.f,0.f};   // A x [t, p, 0...] (cols 0,1 used)
  v8f d2 = {0.f,0.f,0.f,0.f,0.f,0.f,0.f,0.f};   // Gram A x A^T (diag = ||row||^2)

  #pragma unroll 8
  for (int q = 0; q < 16; ++q) {                 // fixed trip: 16 K-chunks per wave
    unsigned kb = (unsigned)(wq + 4*q) * 32u;
    union { v16bf v; v8bf h[2]; } ua;            // A: 16x32 bf16, contiguous k = 16*half + s
    ua.h[0] = *(const v8bf*)(kbase + rbase + kb + cA0);
    ua.h[1] = *(const v8bf*)(kbase + rbase + kb + cA1);
    v16bf bg = *(const v16bf*)(kbase + rbase + kb + cB);   // B = A^T (col n = row n)
    v16bf bt = *(const v16bf*)(bbase + btoff);
    btoff += btstep;
    d1 = __builtin_amdgcn_wmma_f32_16x16x32_bf16(false, ua.v, false, bt, (short)0, d1, false, false);
    d2 = __builtin_amdgcn_wmma_f32_16x16x32_bf16(false, ua.v, false, bg, (short)0, d2, false, false);
  }

  // extract col0 (t-dot), col1 (p-dot) of D1 and diag of D2 per C/D VGPR layout
  #pragma unroll
  for (int r = 0; r < 8; ++r) {
    if (lane == 0)      r_t[group][wq][r]   = d1[r];
    if (lane == 16)     r_t[group][wq][8+r] = d1[r];
    if (lane == 1)      r_p[group][wq][r]   = d1[r];
    if (lane == 17)     r_p[group][wq][8+r] = d1[r];
    if (lane == r)      r_s[group][wq][r]   = d2[r];
    if (lane == 24 + r) r_s[group][wq][8+r] = d2[r];
  }
  __syncthreads();

  if (tid < N_) {                                 // per-leaf gate value
    int g = tid >> 4, l2 = tid & 15;
    float td = 0.f, pd = 0.f, sd = 0.f;
    #pragma unroll
    for (int q = 0; q < 4; ++q) { td += r_t[g][q][l2]; pd += r_p[g][q][l2]; sd += r_s[g][q][l2]; }
    float nn = sqrtf(fmaxf(1.f + 2.f*sc2*pd + sc2*sc2*sd, 0.f));
    float z = (tp + sc2*td) / fmaxf(nn, 1e-12f);
    z *= scale_f;
    sh_h[tid] = 0.5f * z * (1.f + erff(z * 0.70710678118f));  // exact GELU
  }
  __syncthreads();

  // out += sum_n h[n] * mlp2 row n ; thread owns 8 contiguous cols (coalesced b128)
  const __hip_bfloat16* vbase = mlp2 + (((size_t)ii*N_ + jj)*N_)*C_;
  unsigned c0 = (unsigned)tid * 8u;
  float av[8] = {0,0,0,0,0,0,0,0};
  for (int n = 0; n < N_; ++n) {
    float h = sh_h[n];
    v4u u = *(const v4u*)(vbase + (unsigned)n*(unsigned)C_ + c0);
    av[0] += h * lo_bf(u[0]); av[1] += h * hi_bf(u[0]);
    av[2] += h * lo_bf(u[1]); av[3] += h * hi_bf(u[1]);
    av[4] += h * lo_bf(u[2]); av[5] += h * hi_bf(u[2]);
    av[6] += h * lo_bf(u[3]); av[7] += h * hi_bf(u[3]);
  }
  float* dst = acc + (size_t)b*C_ + c0;
  #pragma unroll
  for (int q = 0; q < 8; ++q) unsafeAtomicAdd(dst + q, av[q]);
}

// ============ Phase 4: out = bf16(x + acc) ============
__global__ __launch_bounds__(256) void k_final(
    const __hip_bfloat16* __restrict__ x, const float* __restrict__ acc,
    __hip_bfloat16* __restrict__ out)
{
  int g = blockIdx.x*256 + threadIdx.x;
  if (g < B_*C_) out[g] = __float2bfloat16(bf2f(x[g]) + acc[g]);
}

extern "C" void kernel_launch(void* const* d_in, const int* in_sizes, int n_in,
                              void* d_out, int out_size, void* d_ws, size_t ws_size,
                              hipStream_t stream)
{
  (void)in_sizes; (void)n_in; (void)out_size; (void)ws_size;
  const __hip_bfloat16* x      = (const __hip_bfloat16*)d_in[0];
  const float*          nscale = (const float*)d_in[1];
  const __hip_bfloat16* gate0  = (const __hip_bfloat16*)d_in[2];
  const __hip_bfloat16* gate1  = (const __hip_bfloat16*)d_in[3];
  const __hip_bfloat16* mlp1   = (const __hip_bfloat16*)d_in[4];
  const __hip_bfloat16* mlp2   = (const __hip_bfloat16*)d_in[5];
  const __hip_bfloat16* scale  = (const __hip_bfloat16*)d_in[6];

  char* ws = (char*)d_ws;
  __hip_bfloat16* t_ws = (__hip_bfloat16*)(ws + 0);       // B*C bf16      (16 KB)
  float*          acc  = (float*)(ws + 16384);            // B*C f32       (32 KB)
  int*            idx0 = (int*)(ws + 49152);              // B*K           (256 B)
  int*            idx1 = (int*)(ws + 49664);              // B*K*K         (4 KB)
  float*          tpws = (float*)(ws + 53760);            // B*K*K         (4 KB)
  __hip_bfloat16* p2   = (__hip_bfloat16*)(ws + 57856);   // B*K*K*C bf16  (4 MB)

  k_rms_score<<<B_,        256, 0, stream>>>(x, nscale, gate0, t_ws, idx0, acc);
  k_level1  <<<B_*K_,      256, 0, stream>>>(gate0, gate1, t_ws, idx0, idx1, tpws, p2);
  k_leaf    <<<B_*K_*K_,   256, 0, stream>>>(mlp1, mlp2, t_ws, p2, idx0, idx1, tpws, scale, acc);
  k_final   <<<(B_*C_+255)/256, 256, 0, stream>>>(x, acc, (__hip_bfloat16*)d_out);
}
